// GINModel_26448408609199
// MI455X (gfx1250) — matrix operand order repaired and checked by reading, not compiled, not run on hardware
//
#include <hip/hip_runtime.h>
#include <hip/hip_bf16.h>

// ---------------------------------------------------------------------------
// Problem constants (from the reference)
// ---------------------------------------------------------------------------
#define NN   100000
#define EE   300000
#define BB   2048
#define HH   256
#define NFD  9
#define PDD  640
#define FPDD 1024

// ---------------------------------------------------------------------------
// WMMA fragment types (CDNA5 wave32, V_WMMA_F32_16X16X32_BF16)
// ---------------------------------------------------------------------------
typedef __attribute__((ext_vector_type(16))) __bf16 bf16x16;
typedef __attribute__((ext_vector_type(8)))  float  f32x8;

union FragAB { uint4 u[2]; bf16x16 v; };

__device__ __forceinline__ unsigned int bf16rne(float x) {
    unsigned int u = __float_as_uint(x);
    u += 0x7FFFu + ((u >> 16) & 1u);   // round-to-nearest-even
    return u >> 16;
}
// pack two floats into {hi:bf16(b), lo:bf16(a)}
__device__ __forceinline__ unsigned int pack2bf(float a, float b) {
    return bf16rne(a) | (bf16rne(b) << 16);
}

// ---------------------------------------------------------------------------
// Generic bf16-WMMA GEMM:  C[M,Nc] = epilogue( A[M,K] @ W[K,Nc] + bias )
//   MODE 0: bias
//   MODE 1: bias + ReLU
//   MODE 2: bias + ReLU + BatchNorm (+ optional residual add)
// Requires: K % 32 == 0, Nc % 128 == 0. M arbitrary (row-guarded).
// Block: 256 threads = 8 waves; block tile 128x128; wave tile 64x32.
// ---------------------------------------------------------------------------
template<int MODE>
__global__ __launch_bounds__(256) void gemm_wmma(
    const float* __restrict__ A, const float* __restrict__ W,
    const float* __restrict__ bias, float* __restrict__ C,
    int M, int K, int Nc,
    const float* __restrict__ bn_g, const float* __restrict__ bn_b,
    const float* __restrict__ bn_m, const float* __restrict__ bn_v,
    const float* __restrict__ residual)
{
    __shared__ __align__(16) unsigned short As[128 * 40];  // [row][k], stride 40
    __shared__ __align__(16) unsigned short Ws[128 * 40];  // [n][k] (transposed)

    const int tid  = threadIdx.x;
    const int lane = tid & 31;
    const int wave = tid >> 5;
    const int wm   = wave >> 2;   // 0..1  (64-row slabs)
    const int wn   = wave & 3;    // 0..3  (32-col slabs)
    const int row0 = blockIdx.x * 128;
    const int col0 = blockIdx.y * 128;

    f32x8 acc[4][2];
#pragma unroll
    for (int i = 0; i < 4; ++i)
#pragma unroll
        for (int j = 0; j < 2; ++j)
#pragma unroll
            for (int r = 0; r < 8; ++r) acc[i][j][r] = 0.0f;

    // W-tile cooperative mapping: thread owns column n, contiguous K half.
    const int wN     = tid & 127;        // 0..127
    const int wKhalf = (tid >> 7) * 16;  // 0 or 16

    for (int kk = 0; kk < K; kk += 32) {
        // ---- stage A tile: 128 rows x 32 cols, f32 -> packed bf16 ----
#pragma unroll
        for (int t = 0; t < 4; ++t) {
            int idx  = tid + t * 256;      // 0..1023
            int r    = idx >> 3;
            int c4   = idx & 7;
            int grow = row0 + r;
            float4 f = make_float4(0.f, 0.f, 0.f, 0.f);
            if (grow < M)
                f = *(const float4*)&A[(size_t)grow * K + kk + c4 * 4];
            uint2 h = make_uint2(pack2bf(f.x, f.y), pack2bf(f.z, f.w));
            *(uint2*)&As[r * 40 + c4 * 4] = h;   // ds_store_b64
        }
        // prefetch next A K-slab into cache (global_prefetch_b8)
        if (kk + 32 < K) {
            int prow = row0 + (tid >> 1);
            if (prow < M)
                __builtin_prefetch(&A[(size_t)prow * K + kk + 32], 0, 1);
        }
        // ---- stage W tile transposed: Ws[n][k], 16 contiguous K per thread ----
        {
            unsigned int wv[8];
#pragma unroll
            for (int kq = 0; kq < 8; ++kq) {
                float w0 = W[(size_t)(kk + wKhalf + 2 * kq)     * Nc + col0 + wN];
                float w1 = W[(size_t)(kk + wKhalf + 2 * kq + 1) * Nc + col0 + wN];
                wv[kq] = pack2bf(w0, w1);
            }
            *(uint4*)&Ws[wN * 40 + wKhalf]     = make_uint4(wv[0], wv[1], wv[2], wv[3]);
            *(uint4*)&Ws[wN * 40 + wKhalf + 8] = make_uint4(wv[4], wv[5], wv[6], wv[7]);
        }
        __syncthreads();

        // ---- fragments ----
        // B (32x16): lanes 0-15 -> K=0..15 of col N=lane; lanes 16-31 -> K=16..31
        FragAB bfrag[2];
#pragma unroll
        for (int j = 0; j < 2; ++j) {
            int n  = wn * 32 + j * 16 + (lane & 15);
            int kb = (lane >> 4) * 16;
            const unsigned short* p = &Ws[n * 40 + kb];
            bfrag[j].u[0] = *(const uint4*)p;
            bfrag[j].u[1] = *(const uint4*)(p + 8);
        }
        // A (16x32): lane<16 -> K=0..15; lane>=16 -> K=16..31
        FragAB afrag[4];
#pragma unroll
        for (int i = 0; i < 4; ++i) {
            int r  = wm * 64 + i * 16 + (lane & 15);
            int kb = (lane >> 4) * 16;
            const unsigned short* p = &As[r * 40 + kb];
            afrag[i].u[0] = *(const uint4*)p;
            afrag[i].u[1] = *(const uint4*)(p + 8);
        }
#pragma unroll
        for (int i = 0; i < 4; ++i)
#pragma unroll
            for (int j = 0; j < 2; ++j)
                acc[i][j] = __builtin_amdgcn_wmma_f32_16x16x32_bf16(
                    false, afrag[i].v, false, bfrag[j].v,
                    (short)0, acc[i][j], false, false);
        __syncthreads();
    }

    // ---- epilogue: C layout VGPR r -> M = r + 8*(lane>=16), N = lane&15 ----
#pragma unroll
    for (int i = 0; i < 4; ++i) {
#pragma unroll
        for (int j = 0; j < 2; ++j) {
#pragma unroll
            for (int r = 0; r < 8; ++r) {
                int gm = row0 + wm * 64 + i * 16 + (lane >> 4) * 8 + r;
                int gn = col0 + wn * 32 + j * 16 + (lane & 15);
                if (gm < M) {
                    float v = acc[i][j][r] + bias[gn];
                    if (MODE >= 1) v = fmaxf(v, 0.0f);
                    if (MODE == 2) {
                        v = (v - bn_m[gn]) * rsqrtf(bn_v[gn] + 1e-5f) * bn_g[gn] + bn_b[gn];
                        if (residual) v += residual[(size_t)gm * Nc + gn];
                    }
                    C[(size_t)gm * Nc + gn] = v;
                }
            }
        }
    }
}

// ---------------------------------------------------------------------------
// node_emb.l1 (K=9) + LayerNorm + ReLU, fused. One wave32 per row.
// ---------------------------------------------------------------------------
__global__ __launch_bounds__(256) void nodeemb1_k(
    const float* __restrict__ x, const float* __restrict__ W1,
    const float* __restrict__ b1, const float* __restrict__ g,
    const float* __restrict__ b, float* __restrict__ out, int rows)
{
    int wave = threadIdx.x >> 5, lane = threadIdx.x & 31;
    int row = blockIdx.x * 8 + wave;
    if (row >= rows) return;
    float xr[NFD];
#pragma unroll
    for (int f = 0; f < NFD; ++f) xr[f] = x[(size_t)row * NFD + f];
    float v[8];
    float s = 0.f;
#pragma unroll
    for (int c = 0; c < 8; ++c) {
        int col = lane + 32 * c;
        float acc = b1[col];
#pragma unroll
        for (int f = 0; f < NFD; ++f) acc += xr[f] * W1[f * HH + col];
        v[c] = acc; s += acc;
    }
    for (int m = 16; m >= 1; m >>= 1) s += __shfl_xor(s, m, 32);
    float mu = s * (1.0f / HH);
    float q = 0.f;
#pragma unroll
    for (int c = 0; c < 8; ++c) { float d = v[c] - mu; q += d * d; }
    for (int m = 16; m >= 1; m >>= 1) q += __shfl_xor(q, m, 32);
    float rs = rsqrtf(q * (1.0f / HH) + 1e-5f);
#pragma unroll
    for (int c = 0; c < 8; ++c) {
        int col = lane + 32 * c;
        float y = (v[c] - mu) * rs * g[col] + b[col];
        out[(size_t)row * HH + col] = fmaxf(y, 0.f);
    }
}

// ---------------------------------------------------------------------------
// LayerNorm + ReLU over rows of width D (compile-time). One wave32 per row.
// ---------------------------------------------------------------------------
template<int D>
__global__ __launch_bounds__(256) void lnrelu_k(
    const float* __restrict__ in, float* __restrict__ out,
    const float* __restrict__ g, const float* __restrict__ b, int rows)
{
    constexpr int NC = D / 32;
    int wave = threadIdx.x >> 5, lane = threadIdx.x & 31;
    int row = blockIdx.x * 8 + wave;
    if (row >= rows) return;
    float v[NC];
    float s = 0.f;
#pragma unroll
    for (int c = 0; c < NC; ++c) { v[c] = in[(size_t)row * D + lane + 32 * c]; s += v[c]; }
    for (int m = 16; m >= 1; m >>= 1) s += __shfl_xor(s, m, 32);
    float mu = s * (1.0f / D);
    float q = 0.f;
#pragma unroll
    for (int c = 0; c < NC; ++c) { float d = v[c] - mu; q += d * d; }
    for (int m = 16; m >= 1; m >>= 1) q += __shfl_xor(q, m, 32);
    float rs = rsqrtf(q * (1.0f / D) + 1e-5f);
#pragma unroll
    for (int c = 0; c < NC; ++c) {
        int col = lane + 32 * c;
        float y = (v[c] - mu) * rs * g[col] + b[col];
        out[(size_t)row * D + col] = fmaxf(y, 0.f);
    }
}

// ---------------------------------------------------------------------------
// GIN edge aggregation: accum[dst] += feat[src]  (accum pre-initialized = feat)
// One wave32 per edge, 8 cols/lane, native f32 global atomics.
// ---------------------------------------------------------------------------
__global__ __launch_bounds__(256) void scatter_k(
    const float* __restrict__ feat, const int* __restrict__ src,
    const int* __restrict__ dst, float* __restrict__ accum, int E)
{
    int wave = threadIdx.x >> 5, lane = threadIdx.x & 31;
    int e = blockIdx.x * 8 + wave;
    if (e >= E) return;
    int si = src[e], di = dst[e];
    const float* fp = &feat[(size_t)si * HH];
    float* ap = &accum[(size_t)di * HH];
#pragma unroll
    for (int c = 0; c < 8; ++c) {
        int col = lane + 32 * c;
        unsafeAtomicAdd(&ap[col], fp[col]);
    }
}

// ---------------------------------------------------------------------------
// Batch pooling: sums[bid[row]] += x2[row]; cnt[bid[row]] += 1
// ---------------------------------------------------------------------------
__global__ __launch_bounds__(256) void pool_k(
    const float* __restrict__ x2, const int* __restrict__ bid,
    float* __restrict__ sums, float* __restrict__ cnt, int rows)
{
    int wave = threadIdx.x >> 5, lane = threadIdx.x & 31;
    int row = blockIdx.x * 8 + wave;
    if (row >= rows) return;
    int bb = bid[row];
    const float* xp = &x2[(size_t)row * HH];
    float* sp = &sums[(size_t)bb * HH];
#pragma unroll
    for (int c = 0; c < 8; ++c) {
        int col = lane + 32 * c;
        unsafeAtomicAdd(&sp[col], xp[col]);
    }
    if (lane == 0) unsafeAtomicAdd(&cnt[bb], 1.0f);
}

// mol_emb = 0.5 * (sums + sums / max(cnt,1))
__global__ void molemb_k(const float* __restrict__ sums,
                         const float* __restrict__ cnt,
                         float* __restrict__ memb)
{
    int idx = blockIdx.x * blockDim.x + threadIdx.x;
    if (idx >= BB * HH) return;
    int bb = idx >> 8;
    float c  = fmaxf(cnt[bb], 1.0f);
    float sv = sums[idx];
    memb[idx] = 0.5f * (sv + sv / c);
}

// combined = [mol_emb | pocket_emb | fp_emb]
__global__ void concat_k(const float* __restrict__ m, const float* __restrict__ p,
                         const float* __restrict__ f, float* __restrict__ comb)
{
    int idx = blockIdx.x * blockDim.x + threadIdx.x;
    if (idx >= BB * 3 * HH) return;
    int bb = idx / (3 * HH);
    int c  = idx - bb * (3 * HH);
    float v;
    if (c < HH)            v = m[(size_t)bb * HH + c];
    else if (c < 2 * HH)   v = p[(size_t)bb * HH + (c - HH)];
    else                   v = f[(size_t)bb * HH + (c - 2 * HH)];
    comb[idx] = v;
}

// final [B,128] @ [128,1] + bias. One wave32 per row.
__global__ __launch_bounds__(256) void dense1_k(
    const float* __restrict__ z, const float* __restrict__ W,
    const float* __restrict__ b, float* __restrict__ out, int rows)
{
    int wave = threadIdx.x >> 5, lane = threadIdx.x & 31;
    int row = blockIdx.x * 8 + wave;
    if (row >= rows) return;
    float s = 0.f;
#pragma unroll
    for (int c = 0; c < 4; ++c) {
        int col = lane + 32 * c;
        s += z[(size_t)row * 128 + col] * W[col];
    }
    for (int m = 16; m >= 1; m >>= 1) s += __shfl_xor(s, m, 32);
    if (lane == 0) out[row] = s + b[0];
}

// utility: copy / zero
__global__ void copy4_k(const float4* __restrict__ s, float4* __restrict__ d, long long n4)
{
    long long i = (long long)blockIdx.x * blockDim.x + threadIdx.x;
    long long stride = (long long)gridDim.x * blockDim.x;
    for (; i < n4; i += stride) d[i] = s[i];
}
__global__ void zero_k(float* __restrict__ p, long long n)
{
    long long i = (long long)blockIdx.x * blockDim.x + threadIdx.x;
    long long stride = (long long)gridDim.x * blockDim.x;
    for (; i < n; i += stride) p[i] = 0.f;
}

// ---------------------------------------------------------------------------
// Host-side orchestration
// ---------------------------------------------------------------------------
enum {
    NE1W = 0, NE1B, NELNG, NELNB, NE2W, NE2B,
    C1L1W, C1L1B, C1L2W, C1L2B,
    BN1G, BN1B, BN1M, BN1V,
    C2L1W, C2L1B, C2L2W, C2L2B,
    BN2G, BN2B, BN2M, BN2V,
    PKL1W, PKL1B, PKLNG, PKLNB, PKL2W, PKL2B, PKL3W, PKL3B,
    FPL1W, FPL1B, FPLNG, FPLNB, FPL2W, FPL2B, FPL3W, FPL3B,
    ATL1W, ATL1B, ATL2W, ATL2B,
    PRL1W, PRL1B, PRLNG, PRLNB, PRL2W, PRL2B, PRL3W, PRL3B, PRL4W, PRL4B,
    NPARAMS
};

extern "C" void kernel_launch(void* const* d_in, const int* in_sizes, int n_in,
                              void* d_out, int out_size, void* d_ws, size_t ws_size,
                              hipStream_t stream)
{
    (void)in_sizes; (void)out_size; (void)ws_size;

    const float* x   = (const float*)d_in[0];
    const int*   ei  = (const int*)  d_in[1];
    const int*   bid = (const int*)  d_in[2];
    const float* tf  = (const float*)d_in[3];
    const float* mfp = (const float*)d_in[4];
    const int*   srcIdx = ei;
    const int*   dstIdx = ei + EE;

    const float* P[NPARAMS];
    for (int i = 0; i < NPARAMS && (i + 5) < n_in; ++i)
        P[i] = (const float*)d_in[i + 5];

    // workspace layout (all f32)
    const size_t big = (size_t)NN * HH;
    float* ws    = (float*)d_ws;
    float* bufA  = ws;                  // N x H
    float* bufB  = bufA + big;          // N x H
    float* bufC  = bufB + big;          // N x H
    float* sums  = bufC + big;          // B x H
    float* cnt   = sums + (size_t)BB * HH;            // B
    float* memb  = cnt  + BB;                         // B x H
    float* pemb  = memb + (size_t)BB * HH;            // B x H
    float* femb  = pemb + (size_t)BB * HH;            // B x H
    float* comb  = femb + (size_t)BB * HH;            // B x 3H
    float* t512  = comb + (size_t)BB * 3 * HH;        // B x 512
    float* t512b = t512 + (size_t)BB * 512;           // B x 512
    float* t256  = t512b + (size_t)BB * 512;          // B x 256
    float* t128  = t256 + (size_t)BB * HH;            // B x 128

    const dim3 blk(256);
    const int gN  = (NN + 127) / 128;     // 782 row-blocks for node GEMMs
    const int gB  = (BB + 127) / 128;     // 16 row-blocks for batch GEMMs
    const int gRowWaveN = (NN + 7) / 8;   // wave-per-row over nodes
    const int gRowWaveB = (BB + 7) / 8;   // wave-per-row over batch
    const int gEdge = (EE + 7) / 8;

    // ---- node embedding ----
    nodeemb1_k<<<gRowWaveN, blk, 0, stream>>>(x, P[NE1W], P[NE1B], P[NELNG], P[NELNB], bufA, NN);
    gemm_wmma<0><<<dim3(gN, HH / 128), blk, 0, stream>>>(
        bufA, P[NE2W], P[NE2B], bufB, NN, HH, HH,
        nullptr, nullptr, nullptr, nullptr, nullptr);          // h -> bufB

    // ---- conv1: hplus = h + segment_sum(h[src] -> dst) ----
    copy4_k<<<4096, blk, 0, stream>>>((const float4*)bufB, (float4*)bufC, (long long)(big / 4));
    scatter_k<<<gEdge, blk, 0, stream>>>(bufB, srcIdx, dstIdx, bufC, EE);
    gemm_wmma<1><<<dim3(gN, HH / 128), blk, 0, stream>>>(
        bufC, P[C1L1W], P[C1L1B], bufA, NN, HH, HH,
        nullptr, nullptr, nullptr, nullptr, nullptr);          // t1 -> bufA
    gemm_wmma<2><<<dim3(gN, HH / 128), blk, 0, stream>>>(
        bufA, P[C1L2W], P[C1L2B], bufC, NN, HH, HH,
        P[BN1G], P[BN1B], P[BN1M], P[BN1V], nullptr);          // x1 -> bufC

    // ---- conv2 (+ residual x1) ----
    copy4_k<<<4096, blk, 0, stream>>>((const float4*)bufC, (float4*)bufB, (long long)(big / 4));
    scatter_k<<<gEdge, blk, 0, stream>>>(bufC, srcIdx, dstIdx, bufB, EE);
    gemm_wmma<1><<<dim3(gN, HH / 128), blk, 0, stream>>>(
        bufB, P[C2L1W], P[C2L1B], bufA, NN, HH, HH,
        nullptr, nullptr, nullptr, nullptr, nullptr);
    gemm_wmma<2><<<dim3(gN, HH / 128), blk, 0, stream>>>(
        bufA, P[C2L2W], P[C2L2B], bufB, NN, HH, HH,
        P[BN2G], P[BN2B], P[BN2M], P[BN2V], bufC);             // x2 -> bufB

    // ---- (add+mean)/2 pooling ----
    zero_k<<<1024, blk, 0, stream>>>(sums, (long long)BB * HH + BB);  // sums + cnt
    pool_k<<<gRowWaveN, blk, 0, stream>>>(bufB, bid, sums, cnt, NN);
    molemb_k<<<(BB * HH + 255) / 256, blk, 0, stream>>>(sums, cnt, memb);

    // ---- pocket MLP ----
    gemm_wmma<0><<<dim3(gB, 512 / 128), blk, 0, stream>>>(
        tf, P[PKL1W], P[PKL1B], t512, BB, PDD, 512,
        nullptr, nullptr, nullptr, nullptr, nullptr);
    lnrelu_k<512><<<gRowWaveB, blk, 0, stream>>>(t512, t512b, P[PKLNG], P[PKLNB], BB);
    gemm_wmma<1><<<dim3(gB, HH / 128), blk, 0, stream>>>(
        t512b, P[PKL2W], P[PKL2B], t256, BB, 512, HH,
        nullptr, nullptr, nullptr, nullptr, nullptr);
    gemm_wmma<0><<<dim3(gB, HH / 128), blk, 0, stream>>>(
        t256, P[PKL3W], P[PKL3B], pemb, BB, HH, HH,
        nullptr, nullptr, nullptr, nullptr, nullptr);

    // ---- fingerprint MLP ----
    gemm_wmma<0><<<dim3(gB, 512 / 128), blk, 0, stream>>>(
        mfp, P[FPL1W], P[FPL1B], t512, BB, FPDD, 512,
        nullptr, nullptr, nullptr, nullptr, nullptr);
    lnrelu_k<512><<<gRowWaveB, blk, 0, stream>>>(t512, t512b, P[FPLNG], P[FPLNB], BB);
    gemm_wmma<1><<<dim3(gB, HH / 128), blk, 0, stream>>>(
        t512b, P[FPL2W], P[FPL2B], t256, BB, 512, HH,
        nullptr, nullptr, nullptr, nullptr, nullptr);
    gemm_wmma<0><<<dim3(gB, HH / 128), blk, 0, stream>>>(
        t256, P[FPL3W], P[FPL3B], femb, BB, HH, HH,
        nullptr, nullptr, nullptr, nullptr, nullptr);

    // ---- combined; attention branch is dead code in the reference -> skipped ----
    concat_k<<<(BB * 3 * HH + 255) / 256, blk, 0, stream>>>(memb, pemb, femb, comb);

    // ---- predictor ----
    gemm_wmma<0><<<dim3(gB, 512 / 128), blk, 0, stream>>>(
        comb, P[PRL1W], P[PRL1B], t512, BB, 3 * HH, 512,
        nullptr, nullptr, nullptr, nullptr, nullptr);
    lnrelu_k<512><<<gRowWaveB, blk, 0, stream>>>(t512, t512b, P[PRLNG], P[PRLNB], BB);
    gemm_wmma<1><<<dim3(gB, HH / 128), blk, 0, stream>>>(
        t512b, P[PRL2W], P[PRL2B], t256, BB, 512, HH,
        nullptr, nullptr, nullptr, nullptr, nullptr);
    gemm_wmma<1><<<dim3(gB, 128 / 128), blk, 0, stream>>>(
        t256, P[PRL3W], P[PRL3B], t128, BB, HH, 128,
        nullptr, nullptr, nullptr, nullptr, nullptr);
    dense1_k<<<gRowWaveB, blk, 0, stream>>>(t128, P[PRL4W], P[PRL4B], (float*)d_out, BB);
}
